// DirectPolicy_2602750181699
// MI455X (gfx1250) — matrix operand order repaired and checked by reading, not compiled, not run on hardware
//
#include <hip/hip_runtime.h>
#include <hip/hip_bf16.h>

typedef __attribute__((ext_vector_type(16))) _Float16 v16h;
typedef __attribute__((ext_vector_type(8)))  _Float16 v8h;
typedef __attribute__((ext_vector_type(8)))  float    v8f;

#define BATCH   262144
#define D_U     64
#define K_Y     16
#define HID     200
#define SDIM    18
#define KP1     32            // layer-1 K padded (18 -> 32)
#define NP      208           // hidden cols padded (200 -> 13*16)
#define KP      224           // layers 2/3 K padded (200 -> 7*32)
#define NWAVES  8
#define ROWS_PER_BLOCK (NWAVES * 16)

#define H_ELEMS  (16 * KP)    // 3584 halves per wave activation tile
#define UP_ELEMS (16 * D_U)   // 1024 floats per wave (softplus output)

// Packed-f16 weight workspace layout (bytes), produced by pack_weights_kernel.
// Same [n][k] transposed-padded layout the B-fragment loader expects, so the
// main kernel stages with straight linear async b128 copies.
#define W1_ELEMS (NP * KP1)   // 6656
#define W2_ELEMS (NP * KP)    // 46592
#define W3_ELEMS (D_U * KP)   // 14336
#define WP_TOTAL (W1_ELEMS + W2_ELEMS + W3_ELEMS)   // 67584 halves = 135168 B

// LDS layout (bytes from dynamic-shared base; no static __shared__ used)
#define OFF_W1 0
#define OFF_W2 (OFF_W1 + W1_ELEMS * 2)          // 13312
#define OFF_W3 (OFF_W2 + W2_ELEMS * 2)          // 106496
#define OFF_HA (OFF_W3 + W3_ELEMS * 2)          // 135168
#define OFF_HB (OFF_HA + NWAVES * H_ELEMS * 2)  // 192512
#define OFF_UP (OFF_HB + NWAVES * H_ELEMS * 2)  // 249856
#define SMEM_BYTES (OFF_UP + NWAVES * UP_ELEMS * 4)  // 282624 <= 320KB/WGP

// ---- LDS fragment loaders implementing ISA 7.12.2 wave32 layouts ----

// A fragment (16x32 f16), activations row-major [16][KP].
__device__ __forceinline__ v16h ldsLoadA(const _Float16* hbuf, int lane, int kc) {
  const int row  = lane & 15;
  const int koff = (lane < 16) ? 0 : 8;
  const _Float16* p = hbuf + row * KP + kc + koff;
  v8h lo = *reinterpret_cast<const v8h*>(p);
  v8h hi = *reinterpret_cast<const v8h*>(p + 16);
  return __builtin_shufflevector(lo, hi, 0,1,2,3,4,5,6,7,8,9,10,11,12,13,14,15);
}

// B fragment (32x16 f16), weights transposed Wt[n][k], row stride Kp.
__device__ __forceinline__ v16h ldsLoadB(const _Float16* Wt, int lane, int nb, int kc, int Kp) {
  const int n    = nb + (lane & 15);
  const int koff = (lane < 16) ? 0 : 8;
  const _Float16* p = Wt + n * Kp + kc + koff;
  v8h lo = *reinterpret_cast<const v8h*>(p);
  v8h hi = *reinterpret_cast<const v8h*>(p + 16);
  return __builtin_shufflevector(lo, hi, 0,1,2,3,4,5,6,7,8,9,10,11,12,13,14,15);
}

// D tile (16x16 f32) -> bias + LeakyReLU -> f16 activation buffer [16][KP].
__device__ __forceinline__ void storeTile(_Float16* hbuf, int lane, int nb, v8f acc,
                                          const float* __restrict__ bvec, int nvalid) {
  const int n  = nb + (lane & 15);
  const int m0 = (lane < 16) ? 0 : 8;
  const float bias = (n < nvalid) ? bvec[n] : 0.0f;
#pragma unroll
  for (int i = 0; i < 8; ++i) {
    float v = acc[i] + bias;
    v = (v > 0.0f) ? v : 0.01f * v;          // LeakyReLU slope 0.01
    hbuf[(m0 + i) * KP + n] = (_Float16)v;
  }
}

// ---- Pre-pack kernel: f32 weights -> transposed, padded f16 in workspace ----
__global__ __launch_bounds__(256) void pack_weights_kernel(
    const float* __restrict__ W1, const float* __restrict__ W2,
    const float* __restrict__ W3, _Float16* __restrict__ wp)
{
  int i = blockIdx.x * 256 + threadIdx.x;
  if (i < W1_ELEMS) {                         // region 1: [NP][KP1]
    int n = i / KP1, k = i - n * KP1;
    wp[i] = (k < SDIM && n < HID) ? (_Float16)W1[k * HID + n] : (_Float16)0.0f;
  } else if (i < W1_ELEMS + W2_ELEMS) {       // region 2: [NP][KP]
    int j = i - W1_ELEMS;
    int n = j / KP, k = j - n * KP;
    wp[i] = (k < HID && n < HID) ? (_Float16)W2[k * HID + n] : (_Float16)0.0f;
  } else if (i < WP_TOTAL) {                  // region 3: [D_U][KP]
    int j = i - (W1_ELEMS + W2_ELEMS);
    int n = j / KP, k = j - n * KP;
    wp[i] = (k < HID && n < D_U) ? (_Float16)W3[k * D_U + n] : (_Float16)0.0f;
  }
}

__global__ __launch_bounds__(256) void policy_mlp_wmma_kernel(
    const float* __restrict__ X, const float* __restrict__ Y, const float* __restrict__ TmT,
    const _Float16* __restrict__ wp,                       // packed f16 weights (ws)
    const float* __restrict__ b1, const float* __restrict__ b2, const float* __restrict__ b3,
    float* __restrict__ out)
{
  extern __shared__ char smem[];
  const _Float16* Wt1 = (const _Float16*)(smem + OFF_W1);
  const _Float16* Wt2 = (const _Float16*)(smem + OFF_W2);
  const _Float16* Wt3 = (const _Float16*)(smem + OFF_W3);

  const int tid     = threadIdx.x;
  const int lane    = tid & 31;
  const int wave    = tid >> 5;
  const int rowBase = blockIdx.x * ROWS_PER_BLOCK + wave * 16;

  _Float16* hAw = (_Float16*)(smem + OFF_HA) + wave * H_ELEMS;
  _Float16* hBw = (_Float16*)(smem + OFF_HB) + wave * H_ELEMS;
  float*    upw = (float*)(smem + OFF_UP) + wave * UP_ELEMS;

  // Zero activation buffers (covers K padding 200..223) with b128 stores.
  {
    uint4  zz = make_uint4(0u, 0u, 0u, 0u);
    uint4* z  = (uint4*)(smem + OFF_HA);
    const int tot = (2 * NWAVES * H_ELEMS * 2) / 16;   // 7168 chunks
    for (int i = tid; i < tot; i += 256) z[i] = zz;
  }

  // ---- Stage ALL weights via CDNA5 async global->LDS copies (ASYNCcnt) ----
  {
    const char* src = (const char*)wp;
    const int chunks = (WP_TOTAL * 2) / 16;            // 8448 x 16B
    for (int i = tid; i < chunks; i += 256) {
      uint32_t lo = (uint32_t)(OFF_W1 + (i << 4));     // LDS byte offset (VDST)
      uint32_t go = (uint32_t)(i << 4);                // global byte offset (VADDR, GVS)
      asm volatile("global_load_async_to_lds_b128 %0, %1, %2 offset:0"
                   :: "v"(lo), "v"(go), "s"(src) : "memory");
    }
    asm volatile("s_wait_asynccnt 0x0" ::: "memory");
  }
  __syncthreads();   // the only block-wide barrier: weights visible to all waves

  // ---------------- Layer 1: z(16x18 pad 32) @ W1 -> h1(16x208) ----------------
  v16h a1 = {};
  {
    const int r = rowBase + (lane & 15);
    if (lane < 16) {                       // K = 0..7 and 16..23
      a1[0] = (_Float16)X[r];
#pragma unroll
      for (int j = 0; j < 7; ++j) a1[1 + j] = (_Float16)Y[r * K_Y + j];   // K=1..7
      a1[8] = (_Float16)Y[r * K_Y + 15];   // K=16
      a1[9] = (_Float16)TmT[r];            // K=17 ; 18..23 stay zero
    } else {                               // K = 8..15 ; 24..31 stay zero
#pragma unroll
      for (int j = 0; j < 8; ++j) a1[j] = (_Float16)Y[r * K_Y + 7 + j];
    }
  }
  for (int nt = 0; nt < 13; ++nt) {
    v16h bf = ldsLoadB(Wt1, lane, nt * 16, 0, KP1);
    v8f acc = {};
    acc = __builtin_amdgcn_wmma_f32_16x16x32_f16(false, a1, false, bf, (short)0, acc, false, false);
    storeTile(hAw, lane, nt * 16, acc, b1, HID);
  }

  // ---------------- Layer 2: h1 @ W2 -> h2 ; A fragments cached in VGPRs ----------------
  v16h a2[7];
#pragma unroll
  for (int kk = 0; kk < 7; ++kk) a2[kk] = ldsLoadA(hAw, lane, kk * 32);
  for (int nt = 0; nt < 13; ++nt) {
    v8f acc = {};
#pragma unroll
    for (int kk = 0; kk < 7; ++kk) {
      v16h bf = ldsLoadB(Wt2, lane, nt * 16, kk * 32, KP);
      acc = __builtin_amdgcn_wmma_f32_16x16x32_f16(false, a2[kk], false, bf, (short)0, acc, false, false);
    }
    storeTile(hBw, lane, nt * 16, acc, b2, HID);
  }

  // ---------------- Layer 3: h2 @ W3 -> u_raw(16x64), softplus ----------------
  v16h a3[7];
#pragma unroll
  for (int kk = 0; kk < 7; ++kk) a3[kk] = ldsLoadA(hBw, lane, kk * 32);
  for (int nt = 0; nt < 4; ++nt) {
    v8f acc = {};
#pragma unroll
    for (int kk = 0; kk < 7; ++kk) {
      v16h bf = ldsLoadB(Wt3, lane, nt * 16, kk * 32, KP);
      acc = __builtin_amdgcn_wmma_f32_16x16x32_f16(false, a3[kk], false, bf, (short)0, acc, false, false);
    }
    const int n  = nt * 16 + (lane & 15);
    const int m0 = (lane < 16) ? 0 : 8;
    const float bias = b3[n];
#pragma unroll
    for (int i = 0; i < 8; ++i) {
      float t = 2.0f * (acc[i] + bias);            // softplus(beta*x)/beta, beta=2
      float sp = (t > 20.0f) ? t : ((t < -20.0f) ? __expf(t) : __logf(1.0f + __expf(t)));
      upw[(m0 + i) * D_U + n] = 0.5f * sp;
    }
  }

  // Within-wave LDS RAW fence (lanes are lockstep in wave32; only DScnt matters).
  asm volatile("s_wait_dscnt 0x0" ::: "memory");

  // ---------------- Simplex projection {u>=0, sum u <= 1}: one lane per row ----------------
  if (lane < 16) {
    const int m = lane;
    const int r = rowBase + m;
    const float* src = upw + m * D_U;
    float v[D_U];
    float s = 0.0f;
#pragma unroll 4
    for (int j = 0; j < D_U; ++j) { float x = fmaxf(src[j], 0.0f); v[j] = x; s += x; }
    float4* o4 = (float4*)(out + (size_t)r * D_U);
    if (s <= 1.0f) {
      for (int j = 0; j < 16; ++j)
        o4[j] = make_float4(v[4*j], v[4*j+1], v[4*j+2], v[4*j+3]);
    } else {
      // Sort-free equivalent of reference: extract maxima in descending order,
      // track cumsum; theta = cssv_j / j for the LAST j with u_(j) > cssv_j/j.
      float w[D_U];
      for (int j = 0; j < D_U; ++j) w[j] = v[j];
      float cssv = -1.0f;   // L_CAP = 1.0
      float theta = 0.0f;
      for (int j = 1; j <= D_U; ++j) {
        int mi = 0; float mv = w[0];
        for (int t2 = 1; t2 < D_U; ++t2) { if (w[t2] > mv) { mv = w[t2]; mi = t2; } }
        cssv += mv;
        float th = cssv / (float)j;
        if (mv > th) theta = th;
        w[mi] = -3.0e38f;
      }
      for (int j = 0; j < 16; ++j)
        o4[j] = make_float4(fmaxf(v[4*j+0] - theta, 0.0f), fmaxf(v[4*j+1] - theta, 0.0f),
                            fmaxf(v[4*j+2] - theta, 0.0f), fmaxf(v[4*j+3] - theta, 0.0f));
    }
  }
}

extern "C" void kernel_launch(void* const* d_in, const int* in_sizes, int n_in,
                              void* d_out, int out_size, void* d_ws, size_t ws_size,
                              hipStream_t stream) {
  const float* X   = (const float*)d_in[0];
  const float* Y   = (const float*)d_in[1];
  const float* TmT = (const float*)d_in[2];
  const float* W1  = (const float*)d_in[3];
  const float* b1  = (const float*)d_in[4];
  const float* W2  = (const float*)d_in[5];
  const float* b2  = (const float*)d_in[6];
  const float* W3  = (const float*)d_in[7];
  const float* b3  = (const float*)d_in[8];
  float* out = (float*)d_out;
  _Float16* wp = (_Float16*)d_ws;   // needs 135168 B of scratch

  // 1) pack/transpose/pad weights to f16 once (67584 elems = 264 blocks x 256)
  hipLaunchKernelGGL(pack_weights_kernel, dim3((WP_TOTAL + 255) / 256), dim3(256), 0, stream,
                     W1, W2, W3, wp);

  // 2) fused MLP + projection, all weights LDS-resident, async-staged
  dim3 grid(BATCH / ROWS_PER_BLOCK), block(256);
  hipLaunchKernelGGL(policy_mlp_wmma_kernel, grid, block, (size_t)SMEM_BYTES, stream,
                     X, Y, TmT, wp, b1, b2, b3, out);
}
